// SCAM_6828998001020
// MI455X (gfx1250) — matrix operand order repaired and checked by reading, not compiled, not run on hardware
//
#include <hip/hip_runtime.h>
#include <hip/hip_bf16.h>

typedef _Float16 f16x8 __attribute__((ext_vector_type(8)));
typedef _Float16 v16h  __attribute__((ext_vector_type(16)));
typedef float    v8f   __attribute__((ext_vector_type(8)));

#define Ncst 8
#define Ccst 64
#define Hcst 128
#define Wcst 256
#define HWcst  (Hcst*Wcst)          /* 32768  */
#define CHWcst (Ccst*Hcst*Wcst)     /* 2097152 */

static __device__ __forceinline__ v8f wmma_f16f32(v16h a, v16h b, v8f c) {
  // D = A(16x32 f16) * B(32x16 f16) + C(16x16 f32)
  return __builtin_amdgcn_wmma_f32_16x16x32_f16(false, a, false, b, (short)0, c, false, false);
}

// gfx1250 async global->LDS (ASYNCcnt-tracked, no VGPR round trip).
// LDS byte address = flat_addr[31:0] per ISA 10.2 aperture table.
static __device__ __forceinline__ void async_ld_b128(void* lds, const void* gaddr) {
  asm volatile("global_load_async_to_lds_b128 %0, %1, off"
               :: "v"((unsigned)(unsigned long long)(uintptr_t)lds), "v"(gaddr)
               : "memory");
}
static __device__ __forceinline__ void wait_async0() {
  asm volatile("s_wait_asynccnt 0x0" ::: "memory");
}

// 16-lane xor-butterfly max. Prefer v_permlane16_b32 (pure VALU, no DS/dscnt
// stalls); selector nibbles encode lane i -> lane (i^k) within each 16-lane row.
static __device__ __forceinline__ float rowmax16(float x) {
#if __has_builtin(__builtin_amdgcn_permlane16)
  #define PL16(v, a, b) __builtin_bit_cast(float, __builtin_amdgcn_permlane16( \
      __builtin_bit_cast(unsigned, (v)), __builtin_bit_cast(unsigned, (v)), (a), (b), false, false))
  x = fmaxf(x, PL16(x, 0x67452301u, 0xEFCDAB89u));   // k=1
  x = fmaxf(x, PL16(x, 0x54761032u, 0xDCFE98BAu));   // k=2
  x = fmaxf(x, PL16(x, 0x32107654u, 0xBA98FEDCu));   // k=4
  x = fmaxf(x, PL16(x, 0xFEDCBA98u, 0x76543210u));   // k=8
  #undef PL16
#else
  x = fmaxf(x, __shfl_xor(x, 1));
  x = fmaxf(x, __shfl_xor(x, 2));
  x = fmaxf(x, __shfl_xor(x, 4));
  x = fmaxf(x, __shfl_xor(x, 8));
#endif
  return x;
}

// ---------------------------------------------------------------------------
// Kernel 1: channel-LayerNorm + 1x1 projections.
//   Q_l, Q_r -> f16, layout (N,H,W,C); V_r -> f16, layout (N,H,C,W).
// One block per (n,h) scanline, one thread per w.
// ---------------------------------------------------------------------------
__global__ __launch_bounds__(256) void scam_proj(
    const float* __restrict__ xl, const float* __restrict__ xr,
    const float* __restrict__ nlw, const float* __restrict__ nlb,
    const float* __restrict__ nrw, const float* __restrict__ nrb,
    const float* __restrict__ Wql, const float* __restrict__ bql,
    const float* __restrict__ Wqr, const float* __restrict__ bqr,
    const float* __restrict__ Wvr, const float* __restrict__ bvr,
    _Float16* __restrict__ Ql, _Float16* __restrict__ Qr,
    _Float16* __restrict__ Vr)
{
  __shared__ __align__(16) float sWl[4096], sWr[4096], sWv[4096];
  __shared__ float sp[448];
  const int tid = threadIdx.x;
  // async fill of the three 16KB weight tiles (4 x b128 per thread per tile)
  for (int i = tid; i < 1024; i += 256) {
    async_ld_b128(((uint4*)sWl) + i, ((const uint4*)Wql) + i);
    async_ld_b128(((uint4*)sWr) + i, ((const uint4*)Wqr) + i);
    async_ld_b128(((uint4*)sWv) + i, ((const uint4*)Wvr) + i);
  }
  if (tid < 64) {
    sp[tid]     = nlw[tid]; sp[64+tid]  = nlb[tid];
    sp[128+tid] = nrw[tid]; sp[192+tid] = nrb[tid];
    sp[256+tid] = bql[tid]; sp[320+tid] = bqr[tid];
    sp[384+tid] = bvr[tid];
  }
  wait_async0();
  __syncthreads();

  const int n = blockIdx.x >> 7, h = blockIdx.x & 127, w = tid;
  const size_t gpix = (size_t)n*CHWcst + (size_t)h*Wcst + w;      // + c*HW
  const size_t opix = ((size_t)blockIdx.x*256 + w)*64;            // NHWC

  float xv[64];

  // ------- left: LN + Q_l -------
  #pragma unroll
  for (int c=0;c<64;c++) xv[c] = xl[gpix + (size_t)c*HWcst];
  float mu = 0.f;
  #pragma unroll
  for (int c=0;c<64;c++) mu += xv[c];
  mu *= (1.f/64.f);
  float va = 0.f;
  #pragma unroll
  for (int c=0;c<64;c++){ float d = xv[c]-mu; va += d*d; }
  va *= (1.f/64.f);
  float rstd = rsqrtf(va + 1e-6f);
  #pragma unroll
  for (int c=0;c<64;c++) xv[c] = (xv[c]-mu)*rstd*sp[c] + sp[64+c];
  {
    f16x8* dq = (f16x8*)(Ql + opix);
    for (int og=0; og<8; ++og) {
      float a[8];
      #pragma unroll
      for (int oi=0;oi<8;oi++) a[oi] = sp[256+og*8+oi];
      #pragma unroll
      for (int oi=0;oi<8;oi++){
        const float* wp = sWl + (og*8+oi)*64;
        #pragma unroll
        for (int c=0;c<64;c++) a[oi] += xv[c]*wp[c];
      }
      f16x8 hh;
      #pragma unroll
      for (int oi=0;oi<8;oi++) hh[oi] = (_Float16)a[oi];
      dq[og] = hh;
    }
  }

  // ------- right: V_r from raw, then LN + Q_r -------
  #pragma unroll
  for (int c=0;c<64;c++) xv[c] = xr[gpix + (size_t)c*HWcst];
  {
    const size_t vb = (size_t)blockIdx.x*64*256 + w;              // (N,H,C,W)
    for (int og=0; og<8; ++og) {
      float a[8];
      #pragma unroll
      for (int oi=0;oi<8;oi++) a[oi] = sp[384+og*8+oi];
      #pragma unroll
      for (int oi=0;oi<8;oi++){
        const float* wp = sWv + (og*8+oi)*64;
        #pragma unroll
        for (int c=0;c<64;c++) a[oi] += xv[c]*wp[c];
      }
      #pragma unroll
      for (int oi=0;oi<8;oi++) Vr[vb + (size_t)(og*8+oi)*256] = (_Float16)a[oi];
    }
  }
  mu = 0.f;
  #pragma unroll
  for (int c=0;c<64;c++) mu += xv[c];
  mu *= (1.f/64.f);
  va = 0.f;
  #pragma unroll
  for (int c=0;c<64;c++){ float d = xv[c]-mu; va += d*d; }
  va *= (1.f/64.f);
  rstd = rsqrtf(va + 1e-6f);
  #pragma unroll
  for (int c=0;c<64;c++) xv[c] = (xv[c]-mu)*rstd*sp[128+c] + sp[192+c];
  {
    f16x8* dq = (f16x8*)(Qr + opix);
    for (int og=0; og<8; ++og) {
      float a[8];
      #pragma unroll
      for (int oi=0;oi<8;oi++) a[oi] = sp[320+og*8+oi];
      #pragma unroll
      for (int oi=0;oi<8;oi++){
        const float* wp = sWr + (og*8+oi)*64;
        #pragma unroll
        for (int c=0;c<64;c++) a[oi] += xv[c]*wp[c];
      }
      f16x8 hh;
      #pragma unroll
      for (int oi=0;oi<8;oi++) hh[oi] = (_Float16)a[oi];
      dq[og] = hh;
    }
  }
}

// ---------------------------------------------------------------------------
// Kernel 2: t = x_r - avg_pool3x3(x_r)  (zero pad, always /9)
// ---------------------------------------------------------------------------
__global__ __launch_bounds__(256) void scam_hp(const float* __restrict__ xr,
                                               float* __restrict__ tb)
{
  const int bi = blockIdx.x;
  const int n = bi >> 13, c = (bi >> 7) & 63, h = bi & 127, w = threadIdx.x;
  const size_t base = (size_t)n*CHWcst + (size_t)c*HWcst;
  float s = 0.f;
  #pragma unroll
  for (int dy=-1; dy<=1; ++dy){
    int hh = h + dy;
    if (hh < 0 || hh >= Hcst) continue;
    size_t rb = base + (size_t)hh*Wcst;
    #pragma unroll
    for (int dx=-1; dx<=1; ++dx){
      int ww = w + dx;
      if (ww >= 0 && ww < Wcst) s += xr[rb + ww];
    }
  }
  const size_t gi = base + (size_t)h*Wcst + w;
  tb[gi] = xr[gi] - s * (1.f/9.f);
}

// ---------------------------------------------------------------------------
// Kernel 3: hf = conv3x3(t) + b   (SAME, zero pad). 8 output channels / block.
// ---------------------------------------------------------------------------
__global__ __launch_bounds__(256) void scam_conv3(const float* __restrict__ tb,
    const float* __restrict__ Whf, const float* __restrict__ bhf,
    float* __restrict__ hf)
{
  __shared__ __align__(16) float sw[8*576];
  const int bi = blockIdx.x;
  const int n = bi >> 10, og = (bi >> 7) & 7, h = bi & 127, w = threadIdx.x;
  for (int i = threadIdx.x; i < 8*576; i += 256) sw[i] = Whf[og*8*576 + i];
  __syncthreads();

  float acc[8];
  #pragma unroll
  for (int oi=0;oi<8;oi++) acc[oi] = bhf[og*8+oi];
  const size_t nb = (size_t)n*CHWcst;
  for (int i=0;i<Ccst;i++){
    float xv[9];
    #pragma unroll
    for (int ky=0;ky<3;ky++){
      int hh = h + ky - 1;
      #pragma unroll
      for (int kx=0;kx<3;kx++){
        int ww = w + kx - 1;
        xv[ky*3+kx] = (hh>=0 && hh<Hcst && ww>=0 && ww<Wcst)
            ? tb[nb + (size_t)i*HWcst + (size_t)hh*Wcst + ww] : 0.f;
      }
    }
    #pragma unroll
    for (int oi=0;oi<8;oi++){
      const float* wp = sw + oi*576 + i*9;
      #pragma unroll
      for (int k=0;k<9;k++) acc[oi] += xv[k]*wp[k];
    }
  }
  #pragma unroll
  for (int oi=0;oi<8;oi++)
    hf[nb + (size_t)(og*8+oi)*HWcst + (size_t)h*Wcst + w] = acc[oi];
}

// ---------------------------------------------------------------------------
// Kernel 4: per-scanline flash attention with WMMA + fused final combine.
// One block (8 wave32) per (n,h). Dynamic LDS:
//   sQ [256][64] f16 | sK [256][64] f16 | sV^T [64][256] f16 | sP [8][32][32]
//   sO [256][65] f32 reuses the front after attention.
// ---------------------------------------------------------------------------
__global__ __launch_bounds__(256) void scam_attn(
    const _Float16* __restrict__ Ql, const _Float16* __restrict__ Qr,
    const _Float16* __restrict__ Vr, const float* __restrict__ xl,
    const float* __restrict__ hf, const float* __restrict__ beta,
    const float* __restrict__ gam, float* __restrict__ out)
{
  extern __shared__ char smem[];
  _Float16* sQ = (_Float16*)smem;          // [256][64]
  _Float16* sK = sQ + 256*64;              // [256][64]
  _Float16* sV = sK + 256*64;              // [64][256]  (V transposed: [c][key])
  _Float16* sP = sV + 64*256;              // [8][32][32]
  float*    sO = (float*)smem;             // [256][65]  (reuse after attention)

  const int tid  = threadIdx.x;
  const int wv   = tid >> 5;
  const int lane = tid & 31;
  const int hl   = lane >> 4;              // which 16-lane half
  const int l16  = lane & 15;

  const int bi = blockIdx.x;               // n*128 + h
  const size_t rowoff = (size_t)bi * (256*64);

  // async fill: 3 x 32KB tiles, 8 x b128 per thread per tile
  {
    const uint4* gq = (const uint4*)(Ql + rowoff);
    const uint4* gk = (const uint4*)(Qr + rowoff);
    const uint4* gv = (const uint4*)(Vr + rowoff);
    for (int i = tid; i < 2048; i += 256) {
      async_ld_b128(((uint4*)sQ) + i, gq + i);
      async_ld_b128(((uint4*)sK) + i, gk + i);
      async_ld_b128(((uint4*)sV) + i, gv + i);
    }
    wait_async0();
  }
  __syncthreads();

  _Float16* Pw = sP + wv*32*32;            // per-wave P staging
  const int mbase = wv*32;                 // 32 query rows per wave

  float mst[2][8], lst[2][8];
  v8f acc[2][4];
  #pragma unroll
  for (int t=0;t<2;t++)
    #pragma unroll
    for (int r=0;r<8;r++){ mst[t][r] = -3.0e38f; lst[t][r] = 0.f; }
  #pragma unroll
  for (int t=0;t<2;t++)
    #pragma unroll
    for (int ct=0;ct<4;ct++)
      #pragma unroll
      for (int r=0;r<8;r++) acc[t][ct][r] = 0.f;

  v16h onesb;                              // all-ones B fragment for row sums
  #pragma unroll
  for (int i=0;i<16;i++) onesb[i] = (_Float16)1.0f;

  const float SC = 0.125f;                 // C^-0.5 = 1/8

  for (int j0 = 0; j0 < 256; j0 += 32) {   // 32-key chunks
    #pragma unroll
    for (int t=0;t<2;t++){
      // ---- A fragments: Q rows (16x32 f16 striping per ISA table) ----
      const _Float16* qrow = sQ + (mbase + t*16 + l16)*64;
      v16h a0, a1;
      {
        f16x8 lo = *(const f16x8*)(qrow + hl*8);
        f16x8 hi = *(const f16x8*)(qrow + 16 + hl*8);
        #pragma unroll
        for (int i=0;i<8;i++){ a0[i]=lo[i]; a0[8+i]=hi[i]; }
        lo = *(const f16x8*)(qrow + 32 + hl*8);
        hi = *(const f16x8*)(qrow + 48 + hl*8);
        #pragma unroll
        for (int i=0;i<8;i++){ a1[i]=lo[i]; a1[8+i]=hi[i]; }
      }
      // ---- S = Q K^T over K=64 (2 chained WMMAs per 16x16 N-tile) ----
      v8f s[2];
      #pragma unroll
      for (int nt=0;nt<2;nt++){
        const _Float16* krow = sK + (j0 + nt*16 + l16)*64;
        v16h b0, b1;
        {
          f16x8 lo = *(const f16x8*)(krow + hl*8);
          f16x8 hi = *(const f16x8*)(krow + 16 + hl*8);
          #pragma unroll
          for (int i=0;i<8;i++){ b0[i]=lo[i]; b0[8+i]=hi[i]; }
          lo = *(const f16x8*)(krow + 32 + hl*8);
          hi = *(const f16x8*)(krow + 48 + hl*8);
          #pragma unroll
          for (int i=0;i<8;i++){ b1[i]=lo[i]; b1[8+i]=hi[i]; }
        }
        v8f z;
        #pragma unroll
        for (int i=0;i<8;i++) z[i]=0.f;
        z = wmma_f16f32(a0, b0, z);
        z = wmma_f16f32(a1, b1, z);
        s[nt] = z;
      }
      // ---- online softmax: max via permlane16 butterfly, store P to LDS ----
      float alv[8];
      #pragma unroll
      for (int r=0;r<8;r++){
        float v0 = s[0][r]*SC, v1 = s[1][r]*SC;
        float cm = rowmax16(fmaxf(v0, v1));
        float mo = mst[t][r];
        float mn = fmaxf(mo, cm);
        float al = __expf(mo - mn);
        float p0 = __expf(v0 - mn), p1 = __expf(v1 - mn);
        mst[t][r] = mn;
        alv[r] = al;
        #pragma unroll
        for (int ct=0;ct<4;ct++) acc[t][ct][r] *= al;
        int m = t*16 + hl*8 + r;           // C-layout row -> P scratch row
        Pw[m*32 + l16]      = (_Float16)p0;
        Pw[m*32 + 16 + l16] = (_Float16)p1;
      }
      // ---- P fragment (A-layout) from LDS ----
      const _Float16* prow = Pw + (t*16 + l16)*32;
      v16h pa;
      {
        f16x8 lo = *(const f16x8*)(prow + hl*8);
        f16x8 hi = *(const f16x8*)(prow + 16 + hl*8);
        #pragma unroll
        for (int i=0;i<8;i++){ pa[i]=lo[i]; pa[8+i]=hi[i]; }
      }
      // ---- row sums via WMMA against all-ones B (replaces shuffle tree) ----
      {
        v8f z;
        #pragma unroll
        for (int i=0;i<8;i++) z[i]=0.f;
        v8f rsum = wmma_f16f32(pa, onesb, z);
        #pragma unroll
        for (int r=0;r<8;r++) lst[t][r] = lst[t][r]*alv[r] + rsum[r];
      }
      // ---- O += P V : B = V^T rows (contiguous keys) ----
      #pragma unroll
      for (int ct=0;ct<4;ct++){
        const _Float16* vrow = sV + (ct*16 + l16)*256 + j0 + hl*8;
        v16h vb;
        f16x8 lo = *(const f16x8*)(vrow);
        f16x8 hi = *(const f16x8*)(vrow + 16);
        #pragma unroll
        for (int i=0;i<8;i++){ vb[i]=lo[i]; vb[8+i]=hi[i]; }
        acc[t][ct] = wmma_f16f32(pa, vb, acc[t][ct]);
      }
    }
  }

  __syncthreads();                          // done with sQ/sK/sV; reuse as sO
  #pragma unroll
  for (int t=0;t<2;t++)
    #pragma unroll
    for (int ct=0;ct<4;ct++)
      #pragma unroll
      for (int r=0;r<8;r++){
        int wr = mbase + t*16 + hl*8 + r;
        int c  = ct*16 + l16;
        sO[wr*65 + c] = acc[t][ct][r] / lst[t][r];
      }
  __syncthreads();

  // ---- fused combine: out = x_l + beta*F + gamma*hf (coalesced NCHW) ----
  const int n = bi >> 7, h = bi & 127;
  const size_t pb = (size_t)n*CHWcst + (size_t)h*Wcst + tid;
  for (int c=0;c<Ccst;c++){
    size_t gi = pb + (size_t)c*HWcst;
    out[gi] = xl[gi] + beta[c]*sO[tid*65 + c] + gam[c]*hf[gi];
  }
}

// ---------------------------------------------------------------------------
extern "C" void kernel_launch(void* const* d_in, const int* in_sizes, int n_in,
                              void* d_out, int out_size, void* d_ws, size_t ws_size,
                              hipStream_t stream) {
  const float* x_l  = (const float*)d_in[0];
  const float* x_r  = (const float*)d_in[1];
  const float* nl_w = (const float*)d_in[2];
  const float* nl_b = (const float*)d_in[3];
  const float* nr_w = (const float*)d_in[4];
  const float* nr_b = (const float*)d_in[5];
  const float* Wq_l = (const float*)d_in[6];
  const float* bq_l = (const float*)d_in[7];
  const float* Wq_r = (const float*)d_in[8];
  const float* bq_r = (const float*)d_in[9];
  const float* Wv_r = (const float*)d_in[10];
  const float* bv_r = (const float*)d_in[11];
  const float* beta = (const float*)d_in[12];
  const float* gam  = (const float*)d_in[13];
  const float* W_hf = (const float*)d_in[14];
  const float* b_hf = (const float*)d_in[15];
  float* out = (float*)d_out;

  const size_t NP = (size_t)Ncst*Hcst*Wcst;        // 262144 pixels
  _Float16* Ql = (_Float16*)d_ws;                  // f16 NHWC
  _Float16* Qr = Ql + NP*64;                       // f16 NHWC
  _Float16* Vr = Qr + NP*64;                       // f16 NHCW
  float*    hf = (float*)(Vr + NP*64);             // f32 NCHW
  float*    tb = hf + NP*64;                       // f32 NCHW

  scam_proj <<<Ncst*Hcst,        256, 0,      stream>>>(x_l, x_r, nl_w, nl_b, nr_w, nr_b,
                                                        Wq_l, bq_l, Wq_r, bq_r, Wv_r, bv_r,
                                                        Ql, Qr, Vr);
  scam_hp   <<<Ncst*Ccst*Hcst,   256, 0,      stream>>>(x_r, tb);
  scam_conv3<<<Ncst*8*Hcst,      256, 0,      stream>>>(tb, W_hf, b_hf, hf);
  scam_attn <<<Ncst*Hcst,        256, 114688, stream>>>(Ql, Qr, Vr, x_l, hf, beta, gam, out);
}
